// _CISRJAttentionBlock_87316685128187
// MI455X (gfx1250) — matrix operand order repaired and checked by reading, not compiled, not run on hardware
//
#include <hip/hip_runtime.h>
#include <hip/hip_bf16.h>
#include <math.h>

// ---------------- types ----------------
typedef __bf16 bh8   __attribute__((ext_vector_type(8)));
typedef __bf16 v16bf __attribute__((ext_vector_type(16)));
typedef float  v8f   __attribute__((ext_vector_type(8)));

#define EPS_GN   1e-8f
#define EPS_ATTN 1e-8f

// ---------------- CDNA5 async global->LDS helpers ----------------
// GLOBAL_LOAD_ASYNC_TO_LDS_B128: per-lane DMA of 16B from global to LDS,
// tracked by ASYNCcnt (no VGPR data round-trip). LDS operand is the raw
// 32-bit LDS byte address = low 32 bits of the generic shared pointer.
__device__ __forceinline__ void async_copy_b128(const __bf16* gsrc, __bf16* lds_dst) {
  unsigned lo = (unsigned)(size_t)lds_dst;
  asm volatile("global_load_async_to_lds_b128 %0, %1, off"
               :: "v"(lo), "v"((unsigned long long)(size_t)gsrc)
               : "memory");
}
__device__ __forceinline__ void wait_async0() {
  asm volatile("s_wait_asynccnt 0" ::: "memory");
}

// ---------------- small kernels ----------------

__global__ void f2bf_kernel(const float* __restrict__ in, __bf16* __restrict__ out, int n) {
  int i = blockIdx.x * blockDim.x + threadIdx.x;
  if (i < n) out[i] = (__bf16)in[i];
}

// Stage 1: per-block partial sums of x and x^2 per batch (deterministic two-stage).
__global__ __launch_bounds__(256)
void gn_partial_kernel(const float* __restrict__ x, float* __restrict__ part, int CN) {
  int b = blockIdx.y;
  const float* xb = x + (size_t)b * CN;
  float s = 0.f, s2 = 0.f;
  for (int i = blockIdx.x * blockDim.x + threadIdx.x; i < CN; i += gridDim.x * blockDim.x) {
    float v = xb[i]; s += v; s2 += v * v;
  }
  __shared__ float ls[256], lq[256];
  int t = threadIdx.x;
  ls[t] = s; lq[t] = s2; __syncthreads();
  for (int st = 128; st > 0; st >>= 1) {
    if (t < st) { ls[t] += ls[t + st]; lq[t] += lq[t + st]; }
    __syncthreads();
  }
  if (t == 0) {
    size_t o = ((size_t)b * gridDim.x + blockIdx.x) * 2;
    part[o + 0] = ls[0]; part[o + 1] = lq[0];
  }
}

// Stage 2: reduce 256 partials per batch -> sums[b*2] (sum), sums[b*2+1] (sumsq).
__global__ __launch_bounds__(256)
void gn_final_kernel(const float* __restrict__ part, float* __restrict__ sums) {
  int b = blockIdx.x, t = threadIdx.x;
  float s  = part[((size_t)b * 256 + t) * 2 + 0];
  float s2 = part[((size_t)b * 256 + t) * 2 + 1];
  __shared__ float ls[256], lq[256];
  ls[t] = s; lq[t] = s2; __syncthreads();
  for (int st = 128; st > 0; st >>= 1) {
    if (t < st) { ls[t] += ls[t + st]; lq[t] += lq[t + st]; }
    __syncthreads();
  }
  if (t == 0) { sums[b * 2 + 0] = ls[0]; sums[b * 2 + 1] = lq[0]; }
}

// Normalize (GroupNorm, group=1, affine identity) and depthwise conv k=3 pad=1.
// Writes z_norm f32 [B,C,N] and dwT bf16 [B,N,C].
__global__ __launch_bounds__(256)
void gn_norm_dw_kernel(const float* __restrict__ x, const float* __restrict__ sums,
                       const float* __restrict__ dww,
                       float* __restrict__ zn, __bf16* __restrict__ dwT,
                       int C, int N) {
  size_t idx = (size_t)blockIdx.x * blockDim.x + threadIdx.x;  // over B*C*N
  int n = (int)(idx % N);
  size_t t = idx / N;
  int c = (int)(t % C);
  int b = (int)(t / C);
  float CN   = (float)C * (float)N;
  float mean = sums[b * 2 + 0] / CN;
  float var  = sums[b * 2 + 1] / CN - mean * mean;
  float inv  = rsqrtf(var + EPS_GN);
  const float* xr = x + ((size_t)b * C + c) * N;
  float z  = (xr[n] - mean) * inv;
  zn[idx] = z;
  float zm = (n > 0)     ? (xr[n - 1] - mean) * inv : 0.f;
  float zp = (n < N - 1) ? (xr[n + 1] - mean) * inv : 0.f;
  float dw = dww[c * 3 + 0] * zm + dww[c * 3 + 1] * z + dww[c * 3 + 2] * zp;
  dwT[((size_t)b * N + n) * C + c] = (__bf16)dw;
}

// rowsum[b*N+n] = sum_m s[b][n][m]  (s = relu(logits)^2, bf16)
__global__ __launch_bounds__(256)
void rowsum_kernel(const __bf16* __restrict__ s, float* __restrict__ rs, int N) {
  int row = blockIdx.x;           // b*N + n
  const __bf16* sr = s + (size_t)row * N;
  float acc = 0.f;
  for (int i = threadIdx.x; i < N; i += 256) acc += (float)sr[i];
  __shared__ float ls[256];
  int t = threadIdx.x;
  ls[t] = acc; __syncthreads();
  for (int st = 128; st > 0; st >>= 1) {
    if (t < st) ls[t] += ls[t + st];
    __syncthreads();
  }
  if (t == 0) rs[row] = ls[0];
}

// ---------------- WMMA GEMM ----------------
// D[o,m] = sum_k W[o,k] * X[m,k]   (both operands K-innermost, bf16)
// Block tile 128(O) x 128(M), K-step 32, 8 waves each computing 64x32.
// Double-buffered LDS, async global->LDS DMA overlapped with WMMA.

enum {
  EPI_ZRES_T  = 0,  // + z_norm[o,m] (f32), write transposed bf16 [m][o]
  EPI_SIG_T   = 1,  // sigmoid(+bias[o]),   write transposed bf16
  EPI_BIAS_N  = 2,  // +bias[o],            write natural bf16 [o][m]
  EPI_BIAS_T  = 3,  // +bias[o],            write transposed bf16
  EPI_T       = 4,  // plain,               write transposed bf16
  EPI_RELU2_T = 5,  // relu(acc*scale)^2,   write transposed bf16
  EPI_GATE_T  = 6,  // * 1/(rowsum[m]+eps) * uT[m][o], write transposed bf16
  EPI_RESID_N = 7   // +bias[o] + xres[o][m], write natural f32 [o][m]
};

__device__ __forceinline__ v16bf frag_a(const __bf16* base, int row, int lhi) {
  const bh8* p = reinterpret_cast<const bh8*>(base + row * 32 + lhi * 8);
  bh8 lo = p[0];      // K = lhi*8      .. +7
  bh8 hi = p[2];      // K = 16+lhi*8   .. +7
  return __builtin_shufflevector(lo, hi, 0, 1, 2, 3, 4, 5, 6, 7,
                                 8, 9, 10, 11, 12, 13, 14, 15);
}
__device__ __forceinline__ v16bf frag_b(const __bf16* base, int m, int lhi) {
  const bh8* p = reinterpret_cast<const bh8*>(base + m * 32 + lhi * 8);
  return __builtin_shufflevector(p[0], p[2], 0, 1, 2, 3, 4, 5, 6, 7,
                                 8, 9, 10, 11, 12, 13, 14, 15);
}

template <int EPI>
__global__ __launch_bounds__(256)
void gemm_bf16_kernel(const __bf16* __restrict__ Wg, size_t sWb,
                      const __bf16* __restrict__ Xg, size_t sXb,
                      int O, int M, int K,
                      const float*  __restrict__ bias,
                      const float*  __restrict__ auxF, size_t sAuxF,   // z_norm or x
                      const __bf16* __restrict__ auxB, size_t sAuxB,   // uT
                      const float*  __restrict__ rowsum,
                      float scale,
                      void* __restrict__ outv, size_t sOutb) {
  const int b = blockIdx.z;
  const __bf16* W = Wg + (size_t)b * sWb;
  const __bf16* X = Xg + (size_t)b * sXb;
  const int o_blk = blockIdx.y * 128;
  const int m_blk = blockIdx.x * 128;

  const int lane = threadIdx.x & 31;
  const int wave = threadIdx.x >> 5;     // 8 waves
  const int wo   = wave & 1;             // 2 waves along O  (64 rows each)
  const int wm   = wave >> 1;            // 4 waves along M  (32 cols each)
  const int lrow = lane & 15;
  const int lhi  = lane >> 4;

  __shared__ __align__(16) __bf16 Ws[2][128 * 32];
  __shared__ __align__(16) __bf16 Xs[2][128 * 32];

  v8f acc[4][2] = {};
  const int nk = K >> 5;

  // prologue: DMA slab 0 into buffer 0
#pragma unroll
  for (int it = 0; it < 2; ++it) {
    int i = threadIdx.x + it * 256;      // 512 x 16B per tile
    int row = i >> 2, c4 = i & 3;
    async_copy_b128(W + (size_t)(o_blk + row) * K + c4 * 8, &Ws[0][i * 8]);
    async_copy_b128(X + (size_t)(m_blk + row) * K + c4 * 8, &Xs[0][i * 8]);
  }

  for (int t = 0; t < nk; ++t) {
    const int buf = t & 1;
    wait_async0();        // this wave's DMA into buf complete (LDS visible)
    __syncthreads();      // all waves' DMA done; all waves done reading buf^1

    if (t + 1 < nk) {     // kick off DMA of next slab into the other buffer
      const int kk = (t + 1) * 32;
#pragma unroll
      for (int it = 0; it < 2; ++it) {
        int i = threadIdx.x + it * 256;
        int row = i >> 2, c4 = i & 3;
        async_copy_b128(W + (size_t)(o_blk + row) * K + kk + c4 * 8, &Ws[buf ^ 1][i * 8]);
        async_copy_b128(X + (size_t)(m_blk + row) * K + kk + c4 * 8, &Xs[buf ^ 1][i * 8]);
      }
    }

    v16bf af[4], bfr[2];
#pragma unroll
    for (int ot = 0; ot < 4; ++ot) af[ot] = frag_a(Ws[buf], wo * 64 + ot * 16 + lrow, lhi);
#pragma unroll
    for (int mt = 0; mt < 2; ++mt) bfr[mt] = frag_b(Xs[buf], wm * 32 + mt * 16 + lrow, lhi);

#pragma unroll
    for (int ot = 0; ot < 4; ++ot)
#pragma unroll
      for (int mt = 0; mt < 2; ++mt)
        acc[ot][mt] = __builtin_amdgcn_wmma_f32_16x16x32_bf16(
            false, af[ot], false, bfr[mt], (short)0, acc[ot][mt], false, false);
  }

  // -------- epilogue --------
#pragma unroll
  for (int ot = 0; ot < 4; ++ot) {
#pragma unroll
    for (int mt = 0; mt < 2; ++mt) {
      const int o0 = o_blk + wo * 64 + ot * 16 + lhi * 8;  // 8 consecutive O rows
      const int m  = m_blk + wm * 32 + mt * 16 + lrow;     // this lane's column
      v8f a = acc[ot][mt];
      float val[8];

      if constexpr (EPI == EPI_ZRES_T) {
        const float* zn = auxF + (size_t)b * sAuxF;
#pragma unroll
        for (int i = 0; i < 8; ++i) val[i] = a[i] + zn[(size_t)(o0 + i) * M + m];
      } else if constexpr (EPI == EPI_SIG_T) {
#pragma unroll
        for (int i = 0; i < 8; ++i) {
          float t = a[i] + bias[o0 + i];
          val[i] = 1.f / (1.f + __expf(-t));
        }
      } else if constexpr (EPI == EPI_BIAS_N || EPI == EPI_BIAS_T) {
#pragma unroll
        for (int i = 0; i < 8; ++i) val[i] = a[i] + bias[o0 + i];
      } else if constexpr (EPI == EPI_T) {
#pragma unroll
        for (int i = 0; i < 8; ++i) val[i] = a[i];
      } else if constexpr (EPI == EPI_RELU2_T) {
#pragma unroll
        for (int i = 0; i < 8; ++i) {
          float t = a[i] * scale;
          t = t > 0.f ? t : 0.f;
          val[i] = t * t;
        }
      } else if constexpr (EPI == EPI_GATE_T) {
        float inv = 1.f / (rowsum[(size_t)b * M + m] + EPS_ATTN);
        bh8 ug = *reinterpret_cast<const bh8*>(auxB + (size_t)b * sAuxB + (size_t)m * O + o0);
#pragma unroll
        for (int i = 0; i < 8; ++i) val[i] = a[i] * inv * (float)ug[i];
      } else {  // EPI_RESID_N
        const float* xr = auxF + (size_t)b * sAuxF;
#pragma unroll
        for (int i = 0; i < 8; ++i)
          val[i] = a[i] + bias[o0 + i] + xr[(size_t)(o0 + i) * M + m];
      }

      if constexpr (EPI == EPI_RESID_N) {
        float* of = (float*)outv + (size_t)b * sOutb;
#pragma unroll
        for (int i = 0; i < 8; ++i) of[(size_t)(o0 + i) * M + m] = val[i];
      } else if constexpr (EPI == EPI_BIAS_N) {
        __bf16* ob = (__bf16*)outv + (size_t)b * sOutb;
#pragma unroll
        for (int i = 0; i < 8; ++i) ob[(size_t)(o0 + i) * M + m] = (__bf16)val[i];
      } else {  // transposed bf16 write: [m][o0..o0+7], one contiguous 16B store
        bh8 ov;
#pragma unroll
        for (int i = 0; i < 8; ++i) ov[i] = (__bf16)val[i];
        *reinterpret_cast<bh8*>((__bf16*)outv + (size_t)b * sOutb + (size_t)m * O + o0) = ov;
      }
    }
  }
}

// ---------------- host ----------------

extern "C" void kernel_launch(void* const* d_in, const int* in_sizes, int n_in,
                              void* d_out, int out_size, void* d_ws, size_t ws_size,
                              hipStream_t stream) {
  (void)in_sizes; (void)n_in; (void)out_size; (void)ws_size;

  const float* x    = (const float*)d_in[0];
  const float* dw_w = (const float*)d_in[1];
  const float* pw_w = (const float*)d_in[2];
  const float* u_w  = (const float*)d_in[3];
  const float* u_b  = (const float*)d_in[4];
  const float* v_w  = (const float*)d_in[5];
  const float* v_b  = (const float*)d_in[6];
  const float* h_w  = (const float*)d_in[7];
  const float* h_b  = (const float*)d_in[8];
  const float* q_w  = (const float*)d_in[9];
  const float* k_w  = (const float*)d_in[10];
  const float* o_w  = (const float*)d_in[11];
  const float* o_b  = (const float*)d_in[12];

  constexpr int B = 4, C = 512, N = 2048, C2 = 1024;
  constexpr size_t CN = (size_t)C * N;

  // ---- workspace layout (bytes) ----
  constexpr size_t OFF_SUMS = 0;                                  // B*2 f32
  constexpr size_t OFF_PART = 256;                                // B*256*2 f32
  constexpr size_t OFF_ZN   = 16384;                              // f32  [B,C,N]
  constexpr size_t OFF_ZT   = OFF_ZN  + (size_t)B * CN * 4;       // bf16 [B,N,C]
  constexpr size_t OFF_DWT  = OFF_ZT  + (size_t)B * CN * 2;       // bf16 [B,N,C]
  constexpr size_t OFF_UT   = OFF_DWT + (size_t)B * CN * 2;       // bf16 [B,N,2C]
  constexpr size_t OFF_V    = OFF_UT  + (size_t)B * N * C2 * 2;   // bf16 [B,2C,N]
  constexpr size_t OFF_H    = OFF_V   + (size_t)B * N * C2 * 2;   // bf16 [B,N,C]
  constexpr size_t OFF_Q    = OFF_H   + (size_t)B * CN * 2;       // bf16 [B,N,C]
  constexpr size_t OFF_K    = OFF_Q   + (size_t)B * CN * 2;       // bf16 [B,N,C]
  constexpr size_t OFF_S    = OFF_K   + (size_t)B * CN * 2;       // bf16 [B,N,N]
  constexpr size_t OFF_RS   = OFF_S   + (size_t)B * N * N * 2;    // f32  [B,N]
  constexpr size_t OFF_G    = OFF_RS  + (size_t)B * N * 4;        // bf16 [B,N,2C]
  constexpr size_t OFF_WPW  = OFF_G   + (size_t)B * N * C2 * 2;   // bf16 weights:
  constexpr size_t OFF_WU   = OFF_WPW + (size_t)C * C * 2;
  constexpr size_t OFF_WV   = OFF_WU  + (size_t)C2 * C * 2;
  constexpr size_t OFF_WH   = OFF_WV  + (size_t)C2 * C * 2;
  constexpr size_t OFF_WQ   = OFF_WH  + (size_t)C * C * 2;
  constexpr size_t OFF_WK   = OFF_WQ  + (size_t)C * C * 2;
  constexpr size_t OFF_WO   = OFF_WK  + (size_t)C * C * 2;

  char* w = (char*)d_ws;
  float*  sums = (float*)(w + OFF_SUMS);
  float*  part = (float*)(w + OFF_PART);
  float*  zn   = (float*)(w + OFF_ZN);
  __bf16* zT   = (__bf16*)(w + OFF_ZT);
  __bf16* dwT  = (__bf16*)(w + OFF_DWT);
  __bf16* uT   = (__bf16*)(w + OFF_UT);
  __bf16* vN   = (__bf16*)(w + OFF_V);
  __bf16* hT   = (__bf16*)(w + OFF_H);
  __bf16* qT   = (__bf16*)(w + OFF_Q);
  __bf16* kT   = (__bf16*)(w + OFF_K);
  __bf16* sbf  = (__bf16*)(w + OFF_S);
  float*  rs   = (float*)(w + OFF_RS);
  __bf16* gT   = (__bf16*)(w + OFF_G);
  __bf16* wpw  = (__bf16*)(w + OFF_WPW);
  __bf16* wu   = (__bf16*)(w + OFF_WU);
  __bf16* wv   = (__bf16*)(w + OFF_WV);
  __bf16* wh   = (__bf16*)(w + OFF_WH);
  __bf16* wq   = (__bf16*)(w + OFF_WQ);
  __bf16* wk   = (__bf16*)(w + OFF_WK);
  __bf16* wo   = (__bf16*)(w + OFF_WO);

  // ---- weight conversion f32 -> bf16 ----
  f2bf_kernel<<<(C * C  + 255) / 256, 256, 0, stream>>>(pw_w, wpw, C * C);
  f2bf_kernel<<<(C2 * C + 255) / 256, 256, 0, stream>>>(u_w,  wu,  C2 * C);
  f2bf_kernel<<<(C2 * C + 255) / 256, 256, 0, stream>>>(v_w,  wv,  C2 * C);
  f2bf_kernel<<<(C * C  + 255) / 256, 256, 0, stream>>>(h_w,  wh,  C * C);
  f2bf_kernel<<<(C * C  + 255) / 256, 256, 0, stream>>>(q_w,  wq,  C * C);
  f2bf_kernel<<<(C * C  + 255) / 256, 256, 0, stream>>>(k_w,  wk,  C * C);
  f2bf_kernel<<<(C * C2 + 255) / 256, 256, 0, stream>>>(o_w,  wo,  C * C2);

  // ---- GroupNorm stats (deterministic two-stage) ----
  gn_partial_kernel<<<dim3(256, B), 256, 0, stream>>>(x, part, (int)CN);
  gn_final_kernel<<<B, 256, 0, stream>>>(part, sums);

  // ---- normalize + depthwise conv ----
  gn_norm_dw_kernel<<<(int)((size_t)B * CN / 256), 256, 0, stream>>>(
      x, sums, dw_w, zn, dwT, C, N);

  const float attn_scale = 1.f / sqrtf((float)C);

  // pw: z = z_norm + pw_w @ dw  -> zT [n][c]
  gemm_bf16_kernel<EPI_ZRES_T><<<dim3(N / 128, C / 128, B), 256, 0, stream>>>(
      wpw, 0, dwT, CN, C, N, C, nullptr, zn, CN, nullptr, 0, nullptr, 1.f, zT, CN);

  // u = sigmoid(u_w @ z + u_b)  -> uT [n][2C]
  gemm_bf16_kernel<EPI_SIG_T><<<dim3(N / 128, C2 / 128, B), 256, 0, stream>>>(
      wu, 0, zT, CN, C2, N, C, u_b, nullptr, 0, nullptr, 0, nullptr, 1.f,
      uT, (size_t)N * C2);

  // v = v_w @ z + v_b  -> vN [2C][n] (natural; becomes W of the AV GEMM)
  gemm_bf16_kernel<EPI_BIAS_N><<<dim3(N / 128, C2 / 128, B), 256, 0, stream>>>(
      wv, 0, zT, CN, C2, N, C, v_b, nullptr, 0, nullptr, 0, nullptr, 1.f,
      vN, (size_t)C2 * N);

  // h = (h_w @ z + h_b)^T -> hT [n][c]
  gemm_bf16_kernel<EPI_BIAS_T><<<dim3(N / 128, C / 128, B), 256, 0, stream>>>(
      wh, 0, zT, CN, C, N, C, h_b, nullptr, 0, nullptr, 0, nullptr, 1.f, hT, CN);

  // q = h @ q_w^T -> qT [n][c] ;  k = h @ k_w^T -> kT [n][c]
  gemm_bf16_kernel<EPI_T><<<dim3(N / 128, C / 128, B), 256, 0, stream>>>(
      wq, 0, hT, CN, C, N, C, nullptr, nullptr, 0, nullptr, 0, nullptr, 1.f, qT, CN);
  gemm_bf16_kernel<EPI_T><<<dim3(N / 128, C / 128, B), 256, 0, stream>>>(
      wk, 0, hT, CN, C, N, C, nullptr, nullptr, 0, nullptr, 0, nullptr, 1.f, kT, CN);

  // scores: s[n][m] = relu(scale * q[n]·k[m])^2   (W = k, X = q, transposed write)
  gemm_bf16_kernel<EPI_RELU2_T><<<dim3(N / 128, N / 128, B), 256, 0, stream>>>(
      kT, CN, qT, CN, N, N, C, nullptr, nullptr, 0, nullptr, 0, nullptr,
      attn_scale, sbf, (size_t)N * N);

  // row sums of s
  rowsum_kernel<<<B * N, 256, 0, stream>>>(sbf, rs, N);

  // AV: g[n][o] = u[o][n] * (1/(rowsum[n]+eps)) * sum_m v[o][m] s[n][m]
  gemm_bf16_kernel<EPI_GATE_T><<<dim3(N / 128, C2 / 128, B), 256, 0, stream>>>(
      vN, (size_t)C2 * N, sbf, (size_t)N * N, C2, N, N, nullptr, nullptr, 0,
      uT, (size_t)N * C2, rs, 1.f, gT, (size_t)N * C2);

  // out = x + o_w @ g + o_b   (f32, natural [B,C,N])
  gemm_bf16_kernel<EPI_RESID_N><<<dim3(N / 128, C / 128, B), 256, 0, stream>>>(
      wo, 0, gT, (size_t)N * C2, C, N, C2, o_b, x, CN, nullptr, 0, nullptr, 1.f,
      d_out, CN);
}
